// ChildSumTreeLSTM_33792802685041
// MI455X (gfx1250) — matrix operand order repaired and checked by reading, not compiled, not run on hardware
//
#include <hip/hip_runtime.h>

typedef __bf16 bf16;
typedef __attribute__((ext_vector_type(4)))  float f4;
typedef __attribute__((ext_vector_type(4)))  int   i4;
typedef __attribute__((ext_vector_type(4)))  bf16  v4bf;
typedef __attribute__((ext_vector_type(16))) bf16  v16bf;
typedef __attribute__((ext_vector_type(8)))  float v8f;

union FragAB { v16bf v; i4 q[2]; };
union F8     { f4 v[2]; float f[8]; };

#define DDIM      768
#define KDIM      1536          // stacked [h_l; h_r]
#define MROWS     3840          // [i; o; u; f_l; f_r]
#define BS_STRIDE 1544          // 1536 + 8 bf16 pad (bank spread)

__device__ __forceinline__ float sigf(float x)      { return 1.0f / (1.0f + __expf(-x)); }
__device__ __forceinline__ float tanh_fast(float x) { return 2.0f * sigf(2.0f * x) - 1.0f; }

// Build combined bf16 weight K (3840 x 1536), row-major.
__global__ __launch_bounds__(256) void prep_weights(
    const float* __restrict__ W_iou, const float* __restrict__ U_iou,
    const float* __restrict__ W_f,   const float* __restrict__ U_f,
    bf16* __restrict__ KW)
{
  int idx = blockIdx.x * 256 + threadIdx.x;
  if (idx >= MROWS * KDIM) return;
  int r = idx / KDIM;
  int c = idx - r * KDIM;
  int rightHalf = (c >= DDIM);
  int cc = rightHalf ? (c - DDIM) : c;
  float v;
  if (r < 3 * DDIM) {                       // iou rows: M_iou in both halves
    v = 0.5f * W_iou[r * DDIM + cc] + U_iou[r * DDIM + cc];
  } else if (r < 4 * DDIM) {                // f_l rows: [0.5Wf+Uf | 0.5Wf]
    int rr = r - 3 * DDIM;
    v = 0.5f * W_f[rr * DDIM + cc] + (rightHalf ? 0.0f : U_f[rr * DDIM + cc]);
  } else {                                  // f_r rows: [0.5Wf | 0.5Wf+Uf]
    int rr = r - 4 * DDIM;
    v = 0.5f * W_f[rr * DDIM + cc] + (rightHalf ? U_f[rr * DDIM + cc] : 0.0f);
  }
  KW[idx] = (bf16)v;
}

// One tree level: fused bf16 WMMA GEMM (K @ [h_l;h_r]) + LSTM gates.
// Workgroup = 16 parent columns, 8 waves; wave computes all 5 gate row-groups
// for a 16-wide d-chunk so the epilogue is register-resident.
__global__ __launch_bounds__(256) void tlstm_level(
    const float* __restrict__ h_in, const float* __restrict__ c_in,
    float* __restrict__ h_out, float* __restrict__ c_out,
    const bf16* __restrict__ KW,
    const float* __restrict__ b_iou, const float* __restrict__ b_f,
    int Np, int P)
{
  __shared__ __align__(16) bf16 Bs[16 * BS_STRIDE];
  const int tid  = threadIdx.x;
  const int wgp0 = blockIdx.x * 16;

  // ---- stage B: 16 columns of [h_l; h_r], fp32 -> bf16, column-major in LDS ----
  for (int it = 0; it < 24; ++it) {
    int idx = it * 256 + tid;               // 16 cols * 384 quads = 6144
    int col = idx / 384;
    int kq  = idx - col * 384;
    int k   = kq * 4;
    int p   = wgp0 + col; if (p >= P) p = P - 1;
    int b   = p / Np;
    int nl  = p - b * Np;
    int child = (k >= DDIM) ? 1 : 0;
    int kk  = k - child * DDIM;
    const f4 s = *(const f4*)(h_in + (size_t)(b * (2 * Np) + 2 * nl + child) * DDIM + kk);
    v4bf d = { (bf16)s.x, (bf16)s.y, (bf16)s.z, (bf16)s.w };
    *(v4bf*)(&Bs[col * BS_STRIDE + k]) = d;
  }
  __syncthreads();

  const int lane = tid & 31;
  const int wv   = tid >> 5;
  const int hi   = lane >> 4;                // K sub-block select (WMMA layout)
  const int m    = lane & 15;                // A row within tile / B column

  // Per-lane parent (column m of this workgroup's tile) for the epilogue.
  const int pl = wgp0 + m;
  const int pc = (pl < P) ? pl : (P - 1);
  const int pb = pc / Np;
  const int nl = pc - pb * Np;
  const size_t childBase = (size_t)(pb * (2 * Np) + 2 * nl) * DDIM;
  const size_t outBase   = (size_t)pc * DDIM;
  const bool valid = (pl < P);

  const bf16* __restrict__ bsLane = &Bs[m * BS_STRIDE + hi * 8];

  for (int chunk = wv; chunk < 48; chunk += 8) {   // 48 d-chunks of 16
    const int d0 = chunk * 16;
    v8f acc0 = {0,0,0,0,0,0,0,0};
    v8f acc1 = acc0, acc2 = acc0, acc3 = acc0, acc4 = acc0;

    // A-row base pointers for the 5 gate groups (rows gbase + d0 + m).
    const bf16* a0 = KW + (size_t)(0        + d0 + m) * KDIM + hi * 8;
    const bf16* a1 = KW + (size_t)(DDIM     + d0 + m) * KDIM + hi * 8;
    const bf16* a2 = KW + (size_t)(2 * DDIM + d0 + m) * KDIM + hi * 8;
    const bf16* a3 = KW + (size_t)(3 * DDIM + d0 + m) * KDIM + hi * 8;
    const bf16* a4 = KW + (size_t)(4 * DDIM + d0 + m) * KDIM + hi * 8;

    for (int kt = 0; kt < 48; ++kt) {        // K = 1536 in steps of 32
      const int ko = kt * 32;
      FragAB bf_;
      bf_.q[0] = *(const i4*)(bsLane + ko);
      bf_.q[1] = *(const i4*)(bsLane + ko + 16);
      FragAB af;
      af.q[0] = *(const i4*)(a0 + ko);  af.q[1] = *(const i4*)(a0 + ko + 16);
      acc0 = __builtin_amdgcn_wmma_f32_16x16x32_bf16(false, af.v, false, bf_.v, (short)0, acc0, false, false);
      af.q[0] = *(const i4*)(a1 + ko);  af.q[1] = *(const i4*)(a1 + ko + 16);
      acc1 = __builtin_amdgcn_wmma_f32_16x16x32_bf16(false, af.v, false, bf_.v, (short)0, acc1, false, false);
      af.q[0] = *(const i4*)(a2 + ko);  af.q[1] = *(const i4*)(a2 + ko + 16);
      acc2 = __builtin_amdgcn_wmma_f32_16x16x32_bf16(false, af.v, false, bf_.v, (short)0, acc2, false, false);
      af.q[0] = *(const i4*)(a3 + ko);  af.q[1] = *(const i4*)(a3 + ko + 16);
      acc3 = __builtin_amdgcn_wmma_f32_16x16x32_bf16(false, af.v, false, bf_.v, (short)0, acc3, false, false);
      af.q[0] = *(const i4*)(a4 + ko);  af.q[1] = *(const i4*)(a4 + ko + 16);
      acc4 = __builtin_amdgcn_wmma_f32_16x16x32_bf16(false, af.v, false, bf_.v, (short)0, acc4, false, false);
    }

    // ---- register-resident gate epilogue ----
    const int db = d0 + hi * 8;              // 8 consecutive d per lane (C/D layout)
    F8 bi, bo, bu, bff, clv, crv;
    bi.v[0]  = *(const f4*)(b_iou + db);              bi.v[1]  = *(const f4*)(b_iou + db + 4);
    bo.v[0]  = *(const f4*)(b_iou + DDIM + db);       bo.v[1]  = *(const f4*)(b_iou + DDIM + db + 4);
    bu.v[0]  = *(const f4*)(b_iou + 2 * DDIM + db);   bu.v[1]  = *(const f4*)(b_iou + 2 * DDIM + db + 4);
    bff.v[0] = *(const f4*)(b_f + db);                bff.v[1] = *(const f4*)(b_f + db + 4);
    if (c_in) {
      clv.v[0] = *(const f4*)(c_in + childBase + db);
      clv.v[1] = *(const f4*)(c_in + childBase + db + 4);
      crv.v[0] = *(const f4*)(c_in + childBase + DDIM + db);
      crv.v[1] = *(const f4*)(c_in + childBase + DDIM + db + 4);
    } else {
      f4 z = {0,0,0,0};
      clv.v[0] = z; clv.v[1] = z; crv.v[0] = z; crv.v[1] = z;
    }
    F8 hn, cn;
    #pragma unroll
    for (int j = 0; j < 8; ++j) {
      float iv = sigf(acc0[j] + bi.f[j]);
      float ov = sigf(acc1[j] + bo.f[j]);
      float uv = tanh_fast(acc2[j] + bu.f[j]);
      float fl = sigf(acc3[j] + bff.f[j]);
      float fr = sigf(acc4[j] + bff.f[j]);
      float cc = iv * uv + fl * clv.f[j] + fr * crv.f[j];
      cn.f[j] = cc;
      hn.f[j] = ov * tanh_fast(cc);
    }
    if (valid) {
      *(f4*)(h_out + outBase + db)     = hn.v[0];
      *(f4*)(h_out + outBase + db + 4) = hn.v[1];
      *(f4*)(c_out + outBase + db)     = cn.v[0];
      *(f4*)(c_out + outBase + db + 4) = cn.v[1];
    }
  }
}

extern "C" void kernel_launch(void* const* d_in, const int* in_sizes, int n_in,
                              void* d_out, int out_size, void* d_ws, size_t ws_size,
                              hipStream_t stream)
{
  const float* leaf  = (const float*)d_in[0];
  const float* W_iou = (const float*)d_in[1];
  const float* b_iou = (const float*)d_in[2];
  const float* U_iou = (const float*)d_in[3];
  const float* W_f   = (const float*)d_in[4];
  const float* b_f   = (const float*)d_in[5];
  const float* U_f   = (const float*)d_in[6];

  const int Bb = 8;
  char* ws = (char*)d_ws;
  bf16* KW = (bf16*)ws;
  size_t off = (size_t)MROWS * KDIM * sizeof(bf16);          // 11,796,480 B
  float* h0 = (float*)(ws + off); off += (size_t)Bb * 4096 * DDIM * sizeof(float);
  float* c0 = (float*)(ws + off); off += (size_t)Bb * 4096 * DDIM * sizeof(float);
  float* h1 = (float*)(ws + off); off += (size_t)Bb * 2048 * DDIM * sizeof(float);
  float* c1 = (float*)(ws + off);

  prep_weights<<<(MROWS * KDIM + 255) / 256, 256, 0, stream>>>(W_iou, U_iou, W_f, U_f, KW);

  const float* hin = leaf;
  const float* cin = nullptr;                 // leaves: c = 0
  int N = 8192;
  int bufsel = 0;
  while (N > 1) {
    int Np = N >> 1;
    int P  = Bb * Np;
    float *ho, *co;
    if (Np == 1) { ho = (float*)d_out; co = (float*)d_out + Bb * DDIM; }  // [h_root; c_root]
    else if (bufsel == 0) { ho = h0; co = c0; }
    else                  { ho = h1; co = c1; }
    dim3 grid((P + 15) / 16);
    tlstm_level<<<grid, 256, 0, stream>>>(hin, cin, ho, co, KW, b_iou, b_f, Np, P);
    hin = ho; cin = co;
    N = Np; bufsel ^= 1;
  }
}
